// WindowAttention_13116830122337
// MI455X (gfx1250) — compile-verified, matrix operand order checked by reading, not run on hardware
//
#include <hip/hip_runtime.h>
#include <hip/hip_bf16.h>

typedef __attribute__((ext_vector_type(16))) _Float16 v16h;
typedef __attribute__((ext_vector_type(8)))  _Float16 v8h;
typedef __attribute__((ext_vector_type(4)))  _Float16 v4h;
typedef __attribute__((ext_vector_type(8)))  float    v8f;

#define DIM    128
#define HEADS  4
#define HD     32
#define NTOK   64
#define NWIN   1024
#define XS_STR 136   // 64 x 128 f16 tile stride (halves), 16B aligned rows
#define VT_STR 72    // v-transposed [128][64] stride
#define AS_STR 72    // attn probs [4][64][64] stride
#define QSCALE 0.17677669529663687f   // hd^-0.5

// ---- WMMA operand loader (A-layout: lane holds row lane&15; K pattern
// lanes 0-15 -> K 0..7,16..23 ; lanes 16-31 -> K 8..15,24..31) ----
__device__ __forceinline__ v16h load_tile16(const _Float16* p, int stride) {
    const int lane = threadIdx.x & 31;
    const int row  = lane & 15;
    const int koff = (lane & 16) >> 1;          // 0 or 8
    const _Float16* r = p + row * stride + koff;
    union { v16h v; v8h h[2]; } u;
    u.h[0] = *(const v8h*)(r);
    u.h[1] = *(const v8h*)(r + 16);
    return u.v;
}

__device__ __forceinline__ v8f wmma_f16(v16h a, v16h b, v8f c) {
    return __builtin_amdgcn_wmma_f32_16x16x32_f16(false, a, false, b,
                                                  (short)0, c, false, false);
}

// D-tile (16x16 f32 accum) -> LDS f16 with per-column bias
__device__ __forceinline__ void store_tile_bias(_Float16* p, int stride, v8f acc,
                                                const float* bvec) {
    const int lane = threadIdx.x & 31;
    const int col  = lane & 15;
    const int row0 = (lane & 16) >> 1;
    const float bia = bvec[col];
    #pragma unroll
    for (int r = 0; r < 8; ++r)
        p[(row0 + r) * stride + col] = (_Float16)(acc[r] + bia);
}

// ---- prep kernel 1: gather relative-position bias -> f16 [H][64*64] ----
__global__ void bias_gather_kernel(const float* __restrict__ bias_table,
                                   const int* __restrict__ rel_index,
                                   _Float16* __restrict__ bias_f16) {
    int idx = blockIdx.x * blockDim.x + threadIdx.x;   // H * 4096
    if (idx >= HEADS * NTOK * NTOK) return;
    int h  = idx >> 12;
    int mn = idx & 4095;
    int ri = rel_index[mn];
    bias_f16[h * 4096 + mn] = (_Float16)bias_table[ri * HEADS + h];
}

// ---- prep kernel 2: f32 weights -> f16; fold q-scale into q rows/bias ----
__global__ void wconvert_kernel(const float* __restrict__ qkv_w,
                                const float* __restrict__ qkv_b,
                                const float* __restrict__ proj_w,
                                _Float16* __restrict__ qkvw16,
                                _Float16* __restrict__ projw16,
                                float* __restrict__ qkvb_s) {
    int idx = blockIdx.x * blockDim.x + threadIdx.x;
    if (idx < 3 * DIM * DIM) {
        int row = idx >> 7;                            // 0..383 (q rows < 128)
        float s = (row < DIM) ? QSCALE : 1.0f;
        qkvw16[idx] = (_Float16)(qkv_w[idx] * s);
    } else if (idx < 4 * DIM * DIM) {
        projw16[idx - 3 * DIM * DIM] = (_Float16)proj_w[idx - 3 * DIM * DIM];
    } else if (idx < 4 * DIM * DIM + 3 * DIM) {
        int c = idx - 4 * DIM * DIM;                   // 0..383
        float s = (c < DIM) ? QSCALE : 1.0f;
        qkvb_s[c] = qkv_b[c] * s;
    }
}

// ---- fused window-attention kernel: one block (8 wave32) per window ----
__global__ __launch_bounds__(256)
void window_attn_kernel(const float* __restrict__ x,
                        const float* __restrict__ mask,
                        const _Float16* __restrict__ qkvw16,
                        const float* __restrict__ qkvb_s,
                        const _Float16* __restrict__ projw16,
                        const float* __restrict__ proj_b,
                        const _Float16* __restrict__ bias_f16,
                        float* __restrict__ out) {
    __shared__ _Float16 xs[NTOK * XS_STR];             // x tile, reused as o tile
    __shared__ _Float16 qs[NTOK * XS_STR];             // q (scale folded in W)
    __shared__ _Float16 ks_[NTOK * XS_STR];            // k
    __shared__ _Float16 vt[DIM * VT_STR];              // v transposed: [h*32+d][token]
    __shared__ _Float16 ps[HEADS * NTOK * AS_STR];     // softmax probs

    const int tid  = threadIdx.x;
    const int wave = tid >> 5;
    const int lane = tid & 31;
    const int b    = blockIdx.x;

    // ---- stage x (64x128 f32) -> LDS f16, float4 vectorized ----
    const float4* xb4 = (const float4*)(x + (size_t)b * (NTOK * DIM));
    #pragma unroll
    for (int i = tid; i < NTOK * DIM / 4; i += 256) {
        int e = i * 4;
        int r = e >> 7, c = e & 127;
        float4 f = xb4[i];
        v4h h = { (_Float16)f.x, (_Float16)f.y, (_Float16)f.z, (_Float16)f.w };
        *(v4h*)(xs + r * XS_STR + c) = h;
    }
    __syncthreads();

    // ---- QKV GEMM: (64x128) @ (128x384)^T, uniform store routing ----
    // q tiles (weight rows 0..127)
    #pragma unroll
    for (int t = wave; t < 32; t += 8) {
        int mt = t >> 3, nt = t & 7;
        v8f acc = {};
        #pragma unroll
        for (int kk = 0; kk < 4; ++kk) {
            v16h a  = load_tile16(xs + mt * 16 * XS_STR + kk * 32, XS_STR);
            v16h bm = load_tile16(qkvw16 + nt * 16 * DIM + kk * 32, DIM);
            acc = wmma_f16(a, bm, acc);
        }
        store_tile_bias(qs + mt * 16 * XS_STR + nt * 16, XS_STR, acc,
                        qkvb_s + nt * 16);
    }
    // k tiles (weight rows 128..255)
    #pragma unroll
    for (int t = wave; t < 32; t += 8) {
        int mt = t >> 3, nt = t & 7;
        v8f acc = {};
        #pragma unroll
        for (int kk = 0; kk < 4; ++kk) {
            v16h a  = load_tile16(xs + mt * 16 * XS_STR + kk * 32, XS_STR);
            v16h bm = load_tile16(qkvw16 + (DIM + nt * 16) * DIM + kk * 32, DIM);
            acc = wmma_f16(a, bm, acc);
        }
        store_tile_bias(ks_ + mt * 16 * XS_STR + nt * 16, XS_STR, acc,
                        qkvb_s + DIM + nt * 16);
    }
    // v tiles (weight rows 256..383) -> transposed store into vt
    #pragma unroll
    for (int t = wave; t < 32; t += 8) {
        int mt = t >> 3, nt = t & 7;
        v8f acc = {};
        #pragma unroll
        for (int kk = 0; kk < 4; ++kk) {
            v16h a  = load_tile16(xs + mt * 16 * XS_STR + kk * 32, XS_STR);
            v16h bm = load_tile16(qkvw16 + (2 * DIM + nt * 16) * DIM + kk * 32, DIM);
            acc = wmma_f16(a, bm, acc);
        }
        int col  = nt * 16 + (lane & 15);              // v feature index 0..127
        int row0 = mt * 16 + ((lane & 16) >> 1);       // token index
        float bia = qkvb_s[2 * DIM + nt * 16 + (lane & 15)];
        #pragma unroll
        for (int r = 0; r < 8; ++r)
            vt[col * VT_STR + (row0 + r)] = (_Float16)(acc[r] + bia);
    }
    __syncthreads();

    // ---- attention scores + softmax: 16 row-strips (head, mt), 2 per wave ----
    const float* mrow = mask + (size_t)(b & (NWIN - 1)) * (NTOK * NTOK);
    #pragma unroll
    for (int rt = wave; rt < 16; rt += 8) {
        int h = rt >> 2, mt = rt & 3;
        v16h a = load_tile16(qs + mt * 16 * XS_STR + h * HD, XS_STR);
        v8f acc[4];
        #pragma unroll
        for (int ct = 0; ct < 4; ++ct) {
            v16h bm = load_tile16(ks_ + ct * 16 * XS_STR + h * HD, XS_STR);
            v8f z = {};
            acc[ct] = wmma_f16(a, bm, z);
        }
        // per-lane rows: mt*16 + r + 8*(lane/16); cols: ct*16 + lane%16
        int coll = lane & 15;
        int rof  = (lane & 16) >> 1;
        float vals[4][8];
        #pragma unroll
        for (int ct = 0; ct < 4; ++ct) {
            int n = ct * 16 + coll;
            #pragma unroll
            for (int r = 0; r < 8; ++r) {
                int m = mt * 16 + rof + r;
                vals[ct][r] = acc[ct][r]
                            + (float)bias_f16[h * 4096 + m * 64 + n]
                            + mrow[m * 64 + n];
            }
        }
        #pragma unroll
        for (int r = 0; r < 8; ++r) {
            float mx = vals[0][r];
            #pragma unroll
            for (int ct = 1; ct < 4; ++ct) mx = fmaxf(mx, vals[ct][r]);
            #pragma unroll
            for (int off = 8; off >= 1; off >>= 1)     // reduce within 16-lane half
                mx = fmaxf(mx, __shfl_xor(mx, off, 32));
            float sum = 0.f;
            float e[4];
            #pragma unroll
            for (int ct = 0; ct < 4; ++ct) { e[ct] = __expf(vals[ct][r] - mx); sum += e[ct]; }
            #pragma unroll
            for (int off = 8; off >= 1; off >>= 1)
                sum += __shfl_xor(sum, off, 32);
            float inv = __frcp_rn(sum);
            int m = mt * 16 + rof + r;
            #pragma unroll
            for (int ct = 0; ct < 4; ++ct)
                ps[h * NTOK * AS_STR + m * AS_STR + ct * 16 + coll] =
                    (_Float16)(e[ct] * inv);
        }
    }
    __syncthreads();

    // ---- o = probs @ v : per head (64x64)@(64x32), 32 tiles, 2 k-steps ----
    #pragma unroll
    for (int t = wave; t < 32; t += 8) {
        int h = t >> 3, mt = (t >> 1) & 3, nt = t & 1;
        v8f acc = {};
        #pragma unroll
        for (int kk = 0; kk < 2; ++kk) {
            v16h a  = load_tile16(ps + h * NTOK * AS_STR + mt * 16 * AS_STR + kk * 32, AS_STR);
            v16h bm = load_tile16(vt + (h * HD + nt * 16) * VT_STR + kk * 32, VT_STR);
            acc = wmma_f16(a, bm, acc);
        }
        int col  = h * HD + nt * 16 + (lane & 15);
        int row0 = mt * 16 + ((lane & 16) >> 1);
        #pragma unroll
        for (int r = 0; r < 8; ++r)
            xs[(row0 + r) * XS_STR + col] = (_Float16)acc[r];   // o reuses xs
    }
    __syncthreads();

    // ---- out = o @ proj_w^T + proj_b : 32 tiles of 16x16, 4 k-steps ----
    float* ob = out + (size_t)b * (NTOK * DIM);
    #pragma unroll
    for (int t = wave; t < 32; t += 8) {
        int mt = t >> 3, nt = t & 7;
        v8f acc = {};
        #pragma unroll
        for (int kk = 0; kk < 4; ++kk) {
            v16h a  = load_tile16(xs + mt * 16 * XS_STR + kk * 32, XS_STR);
            v16h bm = load_tile16(projw16 + nt * 16 * DIM + kk * 32, DIM);
            acc = wmma_f16(a, bm, acc);
        }
        int col  = nt * 16 + (lane & 15);
        int row0 = mt * 16 + ((lane & 16) >> 1);
        float bia = proj_b[col];
        #pragma unroll
        for (int r = 0; r < 8; ++r)
            ob[(row0 + r) * DIM + col] = acc[r] + bia;
    }
}

extern "C" void kernel_launch(void* const* d_in, const int* in_sizes, int n_in,
                              void* d_out, int out_size, void* d_ws, size_t ws_size,
                              hipStream_t stream) {
    const float* x          = (const float*)d_in[0];
    const float* mask       = (const float*)d_in[1];
    const float* qkv_w      = (const float*)d_in[2];
    const float* qkv_b      = (const float*)d_in[3];
    const float* proj_w     = (const float*)d_in[4];
    const float* proj_b     = (const float*)d_in[5];
    const float* bias_table = (const float*)d_in[6];
    const int*   rel_index  = (const int*)d_in[7];
    float* out = (float*)d_out;

    // ws layout (f16): bias[4*4096] | qkvw[384*128] | projw[128*128] | f32 qkvb_s[384]
    _Float16* bias_f16 = (_Float16*)d_ws;
    _Float16* qkvw16   = bias_f16 + HEADS * NTOK * NTOK;
    _Float16* projw16  = qkvw16 + 3 * DIM * DIM;
    float*    qkvb_s   = (float*)(projw16 + DIM * DIM);

    bias_gather_kernel<<<(HEADS * 4096 + 255) / 256, 256, 0, stream>>>(
        bias_table, rel_index, bias_f16);
    wconvert_kernel<<<(4 * DIM * DIM + 3 * DIM + 255) / 256, 256, 0, stream>>>(
        qkv_w, qkv_b, proj_w, qkvw16, projw16, qkvb_s);

    const int B = in_sizes[0] / (NTOK * DIM);          // 8192
    window_attn_kernel<<<B, 256, 0, stream>>>(
        x, mask, qkvw16, qkvb_s, projw16, proj_b, bias_f16, out);
}
